// dotproductattention_24833500905696
// MI455X (gfx1250) — matrix operand order repaired and checked by reading, not compiled, not run on hardware
//
#include <hip/hip_runtime.h>
#include <hip/hip_bf16.h>

typedef __attribute__((ext_vector_type(16))) _Float16 v16h;
typedef __attribute__((ext_vector_type(8)))  _Float16 v8h;
typedef __attribute__((ext_vector_type(8)))  float    v8f;
typedef __attribute__((ext_vector_type(4)))  float    v4f;

#define BATCH 16
#define SEQ   2048
#define HDIM  64
#define PSTR  72   // LDS P-stage row stride in halves (144B)

// partner-lane (xor 16) exchange on the VALU pipe: v_permlanex16_b32, identity select
__device__ __forceinline__ float swap16_max(float x) {
    int xi = __builtin_bit_cast(int, x);
    int yi = __builtin_amdgcn_permlanex16(xi, xi, 0x76543210u, 0xFEDCBA98u, false, false);
    return fmaxf(x, __builtin_bit_cast(float, yi));
}

// ---------------------------------------------------------------------------
// Prep: K (fp32 [b][l][d]) -> Kh (f16 [b][l][d]); V (fp32 [b][l][d]) -> Vth (f16 [b][d][l])
// ---------------------------------------------------------------------------
__global__ __launch_bounds__(256)
void attn_prep_kernel(const float* __restrict__ k, const float* __restrict__ v,
                      _Float16* __restrict__ kh, _Float16* __restrict__ vth) {
    int idx = blockIdx.x * blockDim.x + threadIdx.x;   // 0 .. BATCH*SEQ*HDIM-1
    kh[idx] = (_Float16)k[idx];
    int b   = idx / (SEQ * HDIM);
    int rem = idx - b * (SEQ * HDIM);
    int d   = rem / SEQ;
    int l   = rem - d * SEQ;
    vth[idx] = (_Float16)v[((size_t)b * SEQ + l) * HDIM + d];   // coalesced write
}

// ---------------------------------------------------------------------------
// Fused flash attention in transposed orientation:
//   S^T = K A x Q^T B   (lane column = q row -> lane-local softmax)
//   O^T = V^T A x P^T B ; l = 1 A x P^T B
// One wave = 16 query rows; block = 4 waves; grid = BATCH*(SEQ/64) = 512.
// ---------------------------------------------------------------------------
__global__ __launch_bounds__(128)
void attn_fwd_kernel(const float* __restrict__ q,
                     const _Float16* __restrict__ kh,
                     const _Float16* __restrict__ vth,
                     float* __restrict__ out) {
    __shared__ __align__(32) _Float16 plds[4][16 * PSTR];

    const int wave = threadIdx.x >> 5;
    const int lane = threadIdx.x & 31;
    const int n    = lane & 15;        // B/C/D column = q row; A row = key/d row
    const int hi   = lane >> 4;
    const int kb8  = hi * 8;           // 16-bit A-layout K base
    const int kb16 = hi * 16;          // contiguous K base (k = 16*hi + s), used for both A and B here

    const int b  = blockIdx.x >> 5;
    const int qt = blockIdx.x & 31;
    const int q0 = qt * 64 + wave * 16;

    // Q as B fragments of Q^T (lane = q row, halves = contiguous d), log2(e)/8 folded in
    const float QSCALE = 0.125f * 1.44269504088896340736f;
    v16h qb[2];
    const float* qrow = q + ((size_t)(b * SEQ + q0 + n)) * HDIM;
    #pragma unroll
    for (int c = 0; c < 2; ++c) {
        v4f f0 = *(const v4f*)(qrow + c * 32 + kb16);
        v4f f1 = *(const v4f*)(qrow + c * 32 + kb16 + 4);
        v4f f2 = *(const v4f*)(qrow + c * 32 + kb16 + 8);
        v4f f3 = *(const v4f*)(qrow + c * 32 + kb16 + 12);
        #pragma unroll
        for (int i = 0; i < 4; ++i) {
            qb[c][i]      = (_Float16)(QSCALE * f0[i]);
            qb[c][4 + i]  = (_Float16)(QSCALE * f1[i]);
            qb[c][8 + i]  = (_Float16)(QSCALE * f2[i]);
            qb[c][12 + i] = (_Float16)(QSCALE * f3[i]);
        }
    }

    // All-ones A fragment: acc[4] = 1 x P^T -> row sums l, replicated, lane = q row
    v16h ones;
    #pragma unroll
    for (int i = 0; i < 16; ++i) ones[i] = (_Float16)1.0f;

    // acc[0..3] = O^T d-tiles (lane's VGPR r <-> d = 16t + r + 8*hi), acc[4] = l tile
    v8f acc[5];
    float mrow = -1e30f;               // per-lane scalar: running max of this q row
    #pragma unroll
    for (int t = 0; t < 5; ++t) acc[t] = 0.0f;

    const _Float16* kbat = kh  + (size_t)b * SEQ * HDIM;
    const _Float16* vbat = vth + (size_t)b * HDIM * SEQ;
    _Float16* pw = &plds[wave][0];

    for (int kc = 0; kc < SEQ; kc += 64) {
        // ---- S^T = K x Q^T, four 16-key subtiles (log2 domain)
        v8f s[4];
        #pragma unroll
        for (int st = 0; st < 4; ++st) {
            v8f cacc = 0.0f;
            #pragma unroll
            for (int c = 0; c < 2; ++c) {
                const _Float16* kr = kbat + (size_t)(kc + st * 16 + n) * HDIM + c * 32 + kb16;
                v8h a0 = *(const v8h*)(kr);
                v8h a1 = *(const v8h*)(kr + 8);
                v16h ka;
                #pragma unroll
                for (int i = 0; i < 8; ++i) { ka[i] = a0[i]; ka[8 + i] = a1[i]; }
                cacc = __builtin_amdgcn_wmma_f32_16x16x32_f16(
                    false, ka, false, qb[c], (short)0, cacc, false, false);
            }
            s[st] = cacc;
        }

        // ---- lane-local softmax: this lane holds 32 of the 64 scores of q-row n,
        // partner lane (xor 16) holds the other 32 -> one permlanex16 completes the max
        float cm = fmaxf(fmaxf(s[0][0], s[0][1]), fmaxf(s[0][2], s[0][3]));
        cm = fmaxf(cm, fmaxf(fmaxf(s[0][4], s[0][5]), fmaxf(s[0][6], s[0][7])));
        #pragma unroll
        for (int st = 1; st < 4; ++st) {
            float c2 = fmaxf(fmaxf(s[st][0], s[st][1]), fmaxf(s[st][2], s[st][3]));
            c2 = fmaxf(c2, fmaxf(fmaxf(s[st][4], s[st][5]), fmaxf(s[st][6], s[st][7])));
            cm = fmaxf(cm, c2);
        }
        cm = swap16_max(cm);
        float mnew = fmaxf(mrow, cm);
        float scal = __builtin_amdgcn_exp2f(mrow - mnew);
        bool  resc = (mnew != mrow);
        mrow = mnew;
        #pragma unroll
        for (int st = 0; st < 4; ++st)
            #pragma unroll
            for (int r = 0; r < 8; ++r)
                s[st][r] = __builtin_amdgcn_exp2f(s[st][r] - mnew);   // P^T in place

        // wave-uniform guard: skip rescale when no row max moved
        if (__builtin_amdgcn_ballot_w32(resc) != 0u) {
            #pragma unroll
            for (int t = 0; t < 5; ++t)
                #pragma unroll
                for (int r = 0; r < 8; ++r) acc[t][r] *= scal;
        }

        // ---- stage P^T: lane's 8 values per subtile are CONSECUTIVE keys of q-row n
        // -> one packed ds_store_b128 per subtile; reload as B fragments (contiguous)
        #pragma unroll
        for (int st = 0; st < 4; ++st) {
            v8h ph;
            #pragma unroll
            for (int r = 0; r < 8; ++r) ph[r] = (_Float16)s[st][r];
            *(v8h*)(pw + n * PSTR + st * 16 + 8 * hi) = ph;           // [q][key] layout
        }
        v16h pa[2];
        #pragma unroll
        for (int c = 0; c < 2; ++c) {                                 // keys kc+32c ..+31
            v8h lo  = *(const v8h*)(pw + n * PSTR + c * 32 + kb16);
            v8h hi2 = *(const v8h*)(pw + n * PSTR + c * 32 + kb16 + 8);
            #pragma unroll
            for (int i = 0; i < 8; ++i) { pa[c][i] = lo[i]; pa[c][8 + i] = hi2[i]; }
        }

        // ---- O^T += V^T x P^T ; l += 1 x P^T  (all on the matrix pipe)
        #pragma unroll
        for (int c = 0; c < 2; ++c) {
            #pragma unroll
            for (int t = 0; t < 4; ++t) {
                const _Float16* vr = vbat + (size_t)(t * 16 + n) * SEQ + kc + c * 32 + kb16;
                v8h a0 = *(const v8h*)(vr);
                v8h a1 = *(const v8h*)(vr + 8);
                v16h va;
                #pragma unroll
                for (int i = 0; i < 8; ++i) { va[i] = a0[i]; va[8 + i] = a1[i]; }
                acc[t] = __builtin_amdgcn_wmma_f32_16x16x32_f16(
                    false, va, false, pa[c], (short)0, acc[t], false, false);
            }
            acc[4] = __builtin_amdgcn_wmma_f32_16x16x32_f16(
                false, ones, false, pa[c], (short)0, acc[4], false, false);
        }
    }

    // ---- epilogue: O = O^T / l ; lane's 8 acc elements are consecutive d -> b128 stores
    float inv = 1.0f / acc[4][0];
    float* orow = out + ((size_t)(b * SEQ + q0 + n)) * HDIM;
    #pragma unroll
    for (int t = 0; t < 4; ++t) {
        v4f o0, o1;
        #pragma unroll
        for (int i = 0; i < 4; ++i) { o0[i] = acc[t][i] * inv; o1[i] = acc[t][4 + i] * inv; }
        *(v4f*)(orow + t * 16 + 8 * hi)     = o0;
        *(v4f*)(orow + t * 16 + 8 * hi + 4) = o1;
    }
}

// ---------------------------------------------------------------------------
extern "C" void kernel_launch(void* const* d_in, const int* in_sizes, int n_in,
                              void* d_out, int out_size, void* d_ws, size_t ws_size,
                              hipStream_t stream) {
    // setup_inputs order: q, v, k
    const float* q = (const float*)d_in[0];
    const float* v = (const float*)d_in[1];
    const float* k = (const float*)d_in[2];
    float* out = (float*)d_out;

    const size_t nel = (size_t)BATCH * SEQ * HDIM;          // 2,097,152
    _Float16* kh  = (_Float16*)d_ws;                         // 4 MB
    _Float16* vth = (_Float16*)((char*)d_ws + nel * sizeof(_Float16)); // next 4 MB

    attn_prep_kernel<<<(int)(nel / 256), 256, 0, stream>>>(k, v, kh, vth);
    attn_fwd_kernel<<<BATCH * (SEQ / 64), 128, 0, stream>>>(q, kh, vth, out);
}
